// Decoder_4784593567814
// MI455X (gfx1250) — matrix operand order repaired and checked by reading, compiled, not run on hardware
//
#include <hip/hip_runtime.h>
#include <hip/hip_bf16.h>
#include <math.h>

// ---------------- problem constants ----------------
constexpr int Bb = 4;
constexpr int Tt = 4096;
constexpr int Ee = 1024;
constexpr int Hh = 16;
constexpr int Ww = 256;
constexpr int NWw = Tt / Ww;     // 16
constexpr int DHh = Ee / Hh;     // 64
constexpr int Mrows = Bb * Tt;   // 16384

typedef _Float16 v16h __attribute__((ext_vector_type(16)));
typedef _Float16 v8h  __attribute__((ext_vector_type(8)));
typedef float    v8f  __attribute__((ext_vector_type(8)));
typedef unsigned int v4u __attribute__((ext_vector_type(4)));
typedef int      v8i  __attribute__((ext_vector_type(8)));
typedef int      v4i  __attribute__((ext_vector_type(4)));

union H16 { v16h v; v8h h[2]; _Float16 e[16]; };

#define WMMA_F32_F16(a, b, c) \
  __builtin_amdgcn_wmma_f32_16x16x32_f16(false, (a), false, (b), (short)0, (c), false, false)

// Load A fragment (16x32 f16, row-major source, per-lane: row = lane&15,
// K = 16*(lane>>4)+{0..15} in elems 0..15).
__device__ inline v16h load_a_frag(const _Float16* base) {
  H16 u;
  u.h[0] = *(const v8h*)(base);
  u.h[1] = *(const v8h*)(base + 8);
  return u.v;
}

__device__ inline float gelu_new_f(float x) {
  float x3 = x * x * x;
  return 0.5f * x * (1.0f + tanhf(0.7978845608028654f * (x + 0.044715f * x3)));
}

// --------- Tensor Data Mover: 2D f16 tile (global -> LDS), ISA section 8 D# ---------
// tile_d0 elems per row (contiguous), tile_d1 rows, row stride stride_d0 elems.
__device__ inline void tdm_load_2d_f16_to_lds(unsigned lds_off, const void* gaddr,
                                              unsigned tile_d0, unsigned tile_d1,
                                              unsigned tensor_d0, unsigned tensor_d1,
                                              unsigned stride_d0) {
  unsigned long long ga = (unsigned long long)(uintptr_t)gaddr;
  v4u g0;
  g0.x = 1u;                                   // count=1, is_restore=0, gather=0
  g0.y = lds_off;                              // lds_addr (bytes)
  g0.z = (unsigned)(ga & 0xFFFFFFFFu);         // global_addr[31:0]
  g0.w = (unsigned)((ga >> 32) & 0x01FFFFFFu)  // global_addr[56:32]
         | 0x80000000u;                        // type=2 ("image") in [127:126]
  v8i g1;
  g1[0] = (int)(1u << 16);                     // workgroup_mask=0, data_size=1 (2 bytes)
  g1[1] = (int)(tensor_d0 << 16);              // tensor_dim0[15:0] in [63:48]
  g1[2] = (int)((tensor_d0 >> 16) | (tensor_d1 << 16)); // dim0 hi | dim1 lo
  g1[3] = (int)((tensor_d1 >> 16) | (tile_d0 << 16));   // dim1 hi | tile_dim0
  g1[4] = (int)(tile_d1 & 0xFFFFu);            // tile_dim1 | tile_dim2=0
  g1[5] = (int)stride_d0;                      // tensor_dim0_stride[31:0]
  g1[6] = 0;                                   // stride hi | dim1_stride lo
  g1[7] = 0;
  v4i z4 = {0, 0, 0, 0};
#if __clang_major__ >= 23
  v8i z8 = {0, 0, 0, 0, 0, 0, 0, 0};
  __builtin_amdgcn_tensor_load_to_lds(g0, g1, z4, z4, z8, 0);
#else
  __builtin_amdgcn_tensor_load_to_lds(g0, g1, z4, z4, 0);
#endif
}

// ---------------- f32 -> f16 convert ----------------
__global__ __launch_bounds__(256) void cvt_f32_f16(const float* __restrict__ in,
                                                   _Float16* __restrict__ out, int n) {
  int i = (blockIdx.x * 256 + threadIdx.x) * 4;
  if (i + 3 < n) {
    float4 v = *(const float4*)(in + i);
    out[i + 0] = (_Float16)v.x;
    out[i + 1] = (_Float16)v.y;
    out[i + 2] = (_Float16)v.z;
    out[i + 3] = (_Float16)v.w;
  } else {
    for (int j = i; j < n; ++j) out[j] = (_Float16)in[j];
  }
}

// ---------------- LayerNorm (E=1024), f32 in -> f16 out ----------------
__global__ __launch_bounds__(256) void layernorm_f16(const float* __restrict__ x,
                                                     const float* __restrict__ g,
                                                     const float* __restrict__ b,
                                                     _Float16* __restrict__ out) {
  const int row = blockIdx.x;
  const float* xr = x + (size_t)row * Ee;
  float s = 0.f, s2 = 0.f;
  for (int i = threadIdx.x; i < Ee; i += 256) {
    float v = xr[i];
    s += v; s2 += v * v;
  }
  for (int off = 16; off >= 1; off >>= 1) {
    s  += __shfl_xor(s, off);
    s2 += __shfl_xor(s2, off);
  }
  __shared__ float sA[8], sB[8];
  const int wave = threadIdx.x >> 5;
  if ((threadIdx.x & 31) == 0) { sA[wave] = s; sB[wave] = s2; }
  __syncthreads();
  float ts = 0.f, ts2 = 0.f;
#pragma unroll
  for (int w = 0; w < 8; ++w) { ts += sA[w]; ts2 += sB[w]; }
  const float mean = ts * (1.0f / Ee);
  const float var = ts2 * (1.0f / Ee) - mean * mean;
  const float rstd = rsqrtf(var + 1e-5f);
  _Float16* orow = out + (size_t)row * Ee;
  for (int i = threadIdx.x; i < Ee; i += 256)
    orow[i] = (_Float16)((xr[i] - mean) * rstd * g[i] + b[i]);
}

// ---------------- WMMA GEMM: out[M,N] = A[M,K] @ W[N,K]^T + bias (+epilogue) ----------
// EPI 0: f16 store (bias). EPI 1: f32 store (bias + residual). EPI 2: f16 store (gelu(bias+acc)).
template <int EPI>
__global__ __launch_bounds__(256) void gemm_epi(const _Float16* __restrict__ A,
                                                const _Float16* __restrict__ W,
                                                const float* __restrict__ bias,
                                                const float* __restrict__ resid,
                                                void* __restrict__ outp,
                                                int N, int K) {
  const int lane = threadIdx.x & 31;
  const int wave = threadIdx.x >> 5;
  const int half = lane >> 4;
  const int lr = lane & 15;
  const int tiles_n = N >> 6;
  const int tile = blockIdx.x * 8 + wave;
  const int tm = tile / tiles_n;
  const int tn = tile - tm * tiles_n;
  const int mbase = tm << 6;
  const int nbase = tn << 6;

  v8f acc[4][4] = {};

  const _Float16* aptr0 = A + (size_t)(mbase + lr) * K + 16 * half;
  const _Float16* bptr0 = W + (size_t)(nbase + lr) * K + 16 * half;

  for (int k0 = 0; k0 < K; k0 += 32) {
    __builtin_prefetch(aptr0 + k0 + 128, 0, 1);
    __builtin_prefetch(bptr0 + k0 + 128, 0, 1);
    v16h a[4], b[4];
#pragma unroll
    for (int mi = 0; mi < 4; ++mi)
      a[mi] = load_a_frag(aptr0 + (size_t)mi * 16 * K + k0);
#pragma unroll
    for (int ni = 0; ni < 4; ++ni)
      b[ni] = *(const v16h*)(bptr0 + (size_t)ni * 16 * K + k0);
#pragma unroll
    for (int mi = 0; mi < 4; ++mi)
#pragma unroll
      for (int ni = 0; ni < 4; ++ni)
        acc[mi][ni] = WMMA_F32_F16(a[mi], b[ni], acc[mi][ni]);
  }

#pragma unroll
  for (int mi = 0; mi < 4; ++mi) {
#pragma unroll
    for (int ni = 0; ni < 4; ++ni) {
      const int col = nbase + ni * 16 + lr;
      const float bv = bias[col];
      const int row0 = mbase + mi * 16 + 8 * half;
#pragma unroll
      for (int r = 0; r < 8; ++r) {
        const size_t idx = (size_t)(row0 + r) * N + col;
        float v = acc[mi][ni][r] + bv;
        if (EPI == 0) {
          ((_Float16*)outp)[idx] = (_Float16)v;
        } else if (EPI == 1) {
          ((float*)outp)[idx] = resid[idx] + v;
        } else {
          ((_Float16*)outp)[idx] = (_Float16)gelu_new_f(v);
        }
      }
    }
  }
}

// ---------------- windowed causal flash attention --------------------------
// one block per (b, window, head); 8 waves, 32 query rows per wave.
// LDS: sVt [64][256] f16 (32KB) | sK [256][64] f16 (32KB, TDM-loaded) | sP 8x32x32 f16 (16KB)
constexpr unsigned ATTN_LDS = (unsigned)(DHh * Ww * 2 + Ww * DHh * 2 + 8 * 32 * 32 * 2);

__global__ __launch_bounds__(256) void attn_window(const _Float16* __restrict__ qkv,
                                                   _Float16* __restrict__ o) {
  const int h  = blockIdx.x & 15;
  const int nw = (blockIdx.x >> 4) & 15;
  const int b  = blockIdx.x >> 8;
  const size_t row0 = (size_t)b * Tt + (size_t)nw * Ww;

  const _Float16* qbase = qkv + row0 * (3 * Ee) + h * DHh;
  const _Float16* kbase = qbase + Ee;
  const _Float16* vbase = qbase + 2 * Ee;

  extern __shared__ char smem[];
  _Float16* sVt = (_Float16*)smem;                       // [64][256]
  _Float16* sK  = (_Float16*)(smem + DHh * Ww * 2);      // [256][64]
  _Float16* sP  = (_Float16*)(smem + 2 * DHh * Ww * 2);  // 8 x [32][32]

  const int lane = threadIdx.x & 31;
  const int wave = threadIdx.x >> 5;

  // wave0: DMA the K tile (256 rows x 64 halves, row stride 3072 halves) into LDS
  if (wave == 0) {
    tdm_load_2d_f16_to_lds((unsigned)(uintptr_t)sK, (const void*)kbase,
                           /*tile*/ DHh, Ww, /*tensor*/ DHh, Ww, /*stride*/ 3 * Ee);
  }

  // all threads: stage V transposed
  {
    const int t = threadIdx.x;                           // key index 0..255
    const _Float16* vrow = vbase + (size_t)t * (3 * Ee);
#pragma unroll 8
    for (int d = 0; d < DHh; ++d) sVt[d * Ww + t] = vrow[d];
  }
  if (wave == 0) __builtin_amdgcn_s_wait_tensorcnt((short)0);
  __syncthreads();

  const int half = lane >> 4;
  const int lr = lane & 15;
  const int qb = wave * 32;                              // local query base
  _Float16* sPw = sP + wave * (32 * 32);

  // Q fragments (scaled by 1/sqrt(DH) = 0.125)
  v16h qf[2][2];
#pragma unroll
  for (int mt = 0; mt < 2; ++mt)
#pragma unroll
    for (int kk = 0; kk < 2; ++kk) {
      const _Float16* qp = qbase + (size_t)(qb + mt * 16 + lr) * (3 * Ee) + kk * 32 + 16 * half;
      H16 u;
      u.h[0] = *(const v8h*)qp;
      u.h[1] = *(const v8h*)(qp + 8);
#pragma unroll
      for (int e = 0; e < 16; ++e) u.e[e] = (_Float16)((float)u.e[e] * 0.125f);
      qf[mt][kk] = u.v;
    }

  float m_run[2][8], l_run[2][8];
#pragma unroll
  for (int mt = 0; mt < 2; ++mt)
#pragma unroll
    for (int r = 0; r < 8; ++r) { m_run[mt][r] = -1e30f; l_run[mt][r] = 0.f; }
  v8f oacc[2][4] = {};

  for (int kc = 0; kc <= qb; kc += 32) {                 // causal chunk skipping
    // ---- scores: S = (Q/8) K^T for a 32x32 chunk (K from LDS) ----
    v8f sacc[2][2] = {};
#pragma unroll
    for (int nt = 0; nt < 2; ++nt) {
#pragma unroll
      for (int kk = 0; kk < 2; ++kk) {
        const v16h kf = *(const v16h*)(sK + (kc + nt * 16 + lr) * DHh + kk * 32 + 16 * half);
#pragma unroll
        for (int mt = 0; mt < 2; ++mt)
          sacc[mt][nt] = WMMA_F32_F16(qf[mt][kk], kf, sacc[mt][nt]);
      }
    }
    // ---- causal mask: only the diagonal chunk can violate causality ----
    if (kc == qb) {
#pragma unroll
      for (int mt = 0; mt < 2; ++mt)
#pragma unroll
        for (int nt = 0; nt < 2; ++nt) {
          const int key = kc + nt * 16 + lr;
#pragma unroll
          for (int r = 0; r < 8; ++r) {
            const int qrow = qb + mt * 16 + r + 8 * half;
            if (key > qrow) sacc[mt][nt][r] = -1e30f;
          }
        }
    }
    // ---- online softmax (stage-major butterflies: 8 independent bpermutes per wait) ----
#pragma unroll
    for (int mt = 0; mt < 2; ++mt) {
      float cm[8], corr_[8], ps[8];
#pragma unroll
      for (int r = 0; r < 8; ++r)
        cm[r] = fmaxf(sacc[mt][0][r], sacc[mt][1][r]);
#pragma unroll
      for (int st = 1; st <= 8; st <<= 1) {
#pragma unroll
        for (int r = 0; r < 8; ++r)
          cm[r] = fmaxf(cm[r], __shfl_xor(cm[r], st, 16));
      }
#pragma unroll
      for (int r = 0; r < 8; ++r) {
        const float mold = m_run[mt][r];
        const float mnew = fmaxf(mold, cm[r]);
        corr_[r] = __expf(mold - mnew);
        m_run[mt][r] = mnew;
        const float p0 = __expf(sacc[mt][0][r] - mnew);
        const float p1 = __expf(sacc[mt][1][r] - mnew);
        ps[r] = p0 + p1;
        const int mloc = mt * 16 + r + 8 * half;
        sPw[mloc * 32 + lr] = (_Float16)p0;
        sPw[mloc * 32 + 16 + lr] = (_Float16)p1;
      }
#pragma unroll
      for (int st = 1; st <= 8; st <<= 1) {
#pragma unroll
        for (int r = 0; r < 8; ++r)
          ps[r] += __shfl_xor(ps[r], st, 16);
      }
#pragma unroll
      for (int r = 0; r < 8; ++r)
        l_run[mt][r] = l_run[mt][r] * corr_[r] + ps[r];
#pragma unroll
      for (int nt = 0; nt < 4; ++nt)
#pragma unroll
        for (int r = 0; r < 8; ++r)
          oacc[mt][nt][r] *= corr_[r];
    }
    asm volatile("s_wait_dscnt 0" ::: "memory");
    // ---- O += P @ V ----
#pragma unroll
    for (int mt = 0; mt < 2; ++mt) {
      const _Float16* pp = sPw + (mt * 16 + lr) * 32 + 16 * half;
      H16 u;
      u.h[0] = *(const v8h*)pp;
      u.h[1] = *(const v8h*)(pp + 8);
      const v16h pa = u.v;
#pragma unroll
      for (int nt = 0; nt < 4; ++nt) {
        const v16h vb = *(const v16h*)(sVt + (nt * 16 + lr) * Ww + kc + 16 * half);
        oacc[mt][nt] = WMMA_F32_F16(pa, vb, oacc[mt][nt]);
      }
    }
  }

  // ---- normalize + store f16 ----
#pragma unroll
  for (int mt = 0; mt < 2; ++mt)
#pragma unroll
    for (int r = 0; r < 8; ++r) {
      const float inv = 1.0f / l_run[mt][r];
      const size_t trow = row0 + qb + mt * 16 + r + 8 * half;
#pragma unroll
      for (int nt = 0; nt < 4; ++nt)
        o[trow * Ee + h * DHh + nt * 16 + lr] = (_Float16)(oacc[mt][nt][r] * inv);
    }
}

// ---------------- host-side orchestration ----------------
extern "C" void kernel_launch(void* const* d_in, const int* in_sizes, int n_in,
                              void* d_out, int out_size, void* d_ws, size_t ws_size,
                              hipStream_t stream) {
  (void)in_sizes; (void)n_in; (void)out_size; (void)ws_size;
  const float* x     = (const float*)d_in[0];
  const float* ln1_g = (const float*)d_in[1];
  const float* ln1_b = (const float*)d_in[2];
  const float* ln2_g = (const float*)d_in[3];
  const float* ln2_b = (const float*)d_in[4];
  const float* w_in  = (const float*)d_in[5];
  const float* b_in  = (const float*)d_in[6];
  const float* w_out = (const float*)d_in[7];
  const float* b_out = (const float*)d_in[8];
  const float* w1    = (const float*)d_in[9];
  const float* b1    = (const float*)d_in[10];
  const float* w2    = (const float*)d_in[11];
  const float* b2    = (const float*)d_in[12];
  float* out = (float*)d_out;

  char* p = (char*)d_ws;
  auto take = [&](size_t bytes) -> char* {
    char* r = p;
    p += (bytes + 255) & ~(size_t)255;
    return r;
  };
  const size_t EE = (size_t)Ee * Ee;                     // 1M
  _Float16* wInH  = (_Float16*)take(3 * EE * 2);
  _Float16* wOutH = (_Float16*)take(EE * 2);
  _Float16* w1H   = (_Float16*)take(4 * EE * 2);
  _Float16* w2H   = (_Float16*)take(4 * EE * 2);
  _Float16* xln   = (_Float16*)take((size_t)Mrows * Ee * 2);
  _Float16* qkvH  = (_Float16*)take((size_t)Mrows * 3 * Ee * 2);
  _Float16* attnO = (_Float16*)take((size_t)Mrows * Ee * 2);
  float*    x2    = (float*)   take((size_t)Mrows * Ee * 4);
  _Float16* hln   = (_Float16*)take((size_t)Mrows * Ee * 2);
  _Float16* actH  = (_Float16*)take((size_t)Mrows * 4 * Ee * 2);

  // weight conversion (f32 -> f16)
  cvt_f32_f16<<<(int)(3 * EE / 1024), 256, 0, stream>>>(w_in, wInH, (int)(3 * EE));
  cvt_f32_f16<<<(int)(EE / 1024),     256, 0, stream>>>(w_out, wOutH, (int)EE);
  cvt_f32_f16<<<(int)(4 * EE / 1024), 256, 0, stream>>>(w1, w1H, (int)(4 * EE));
  cvt_f32_f16<<<(int)(4 * EE / 1024), 256, 0, stream>>>(w2, w2H, (int)(4 * EE));

  // LN1
  layernorm_f16<<<Mrows, 256, 0, stream>>>(x, ln1_g, ln1_b, xln);

  // QKV projection: [16384,1024] x [3072,1024]^T -> f16 [16384,3072]
  {
    int tiles = (Mrows / 64) * (3 * Ee / 64);
    gemm_epi<0><<<tiles / 8, 256, 0, stream>>>(xln, wInH, b_in, nullptr, qkvH, 3 * Ee, Ee);
  }

  // windowed attention (dynamic LDS: 80KB)
  attn_window<<<Bb * NWw * Hh, 256, ATTN_LDS, stream>>>(qkvH, attnO);

  // out projection + residual: x2 = x + attnO @ w_out^T + b_out  (f32)
  {
    int tiles = (Mrows / 64) * (Ee / 64);
    gemm_epi<1><<<tiles / 8, 256, 0, stream>>>(attnO, wOutH, b_out, x, x2, Ee, Ee);
  }

  // LN2
  layernorm_f16<<<Mrows, 256, 0, stream>>>(x2, ln2_g, ln2_b, hln);

  // MLP fc1 + gelu -> f16 act
  {
    int tiles = (Mrows / 64) * (4 * Ee / 64);
    gemm_epi<2><<<tiles / 8, 256, 0, stream>>>(hln, w1H, b1, nullptr, actH, 4 * Ee, Ee);
  }

  // MLP fc2 + residual -> f32 out
  {
    int tiles = (Mrows / 64) * (Ee / 64);
    gemm_epi<1><<<tiles / 8, 256, 0, stream>>>(actH, w2H, b2, x2, out, Ee, 4 * Ee);
  }
}
